// PatchcoreModel_58780922413476
// MI455X (gfx1250) — compile-verified, matrix operand order checked by reading, not run on hardware
//
#include <hip/hip_runtime.h>
#include <math.h>

// ---------------------------------------------------------------------------
// PatchCore inference pipeline for MI455X (gfx1250).
// Dominant op: 5832x32000 L2-distance search over K=1536 (~573 GFLOP GEMM)
// done with V_WMMA_F32_16X16X32_BF16 + fused register-resident min/argmin.
// memory_bank converted once to bf16 (98.3 MB) => fully L2-resident (192 MB).
// Main kernel is software-pipelined (register double-buffered fragments) and
// each wave computes a 16x32 slab (A-fragment reused across two WMMAs).
// Workspace requirement: ~127 MB.
// ---------------------------------------------------------------------------

typedef __attribute__((ext_vector_type(16))) __bf16 bf16x16;
typedef __attribute__((ext_vector_type(8)))  float   f32x8;
typedef __attribute__((ext_vector_type(4)))  unsigned int u32x4;

union FragAB { bf16x16 v; u32x4 q[2]; };

// ---- constants --------------------------------------------------------------
#define NB      8        // batch
#define HP      27       // pooled spatial
#define PP      729      // 27*27
#define ROWS    5832     // NB*PP
#define ROWSP   5888     // padded to 92*64
#define KC      1536     // embedding channels
#define MBANK   32000    // memory bank rows
#define NSEG    5        // N-segments for the big kernel
#define SEGLEN  6400     // MBANK / NSEG, divisible by 64
#define LDS_STRIDE 1544  // 1536 + 8 pad (bank-conflict-free b128 LDS reads)

__device__ __forceinline__ unsigned short f2bf(float f) {
    union { float f; unsigned u; } v; v.f = f;
    unsigned u = v.u;
    u += 0x7FFFu + ((u >> 16) & 1u);   // round-to-nearest-even
    return (unsigned short)(u >> 16);
}
__device__ __forceinline__ float bf2f(unsigned short h) {
    union { unsigned u; float f; } v; v.u = ((unsigned)h) << 16;
    return v.f;
}
__device__ __forceinline__ int reflect_idx(int i, int n) {
    if (i < 0) i = -i;
    if (i >= n) i = 2 * n - 2 - i;
    return i;
}

// ---- 1. avgpool(4,1,1) on layer3 [8,1024,14,14] -> [8,1024,13,13] ----------
__global__ __launch_bounds__(256) void k_pool_f3(const float* __restrict__ f3,
                                                 float* __restrict__ f3p) {
    int idx = blockIdx.x * 256 + threadIdx.x;
    const int total = NB * 1024 * 13 * 13;
    if (idx >= total) return;
    int x = idx % 13, y = (idx / 13) % 13, bc = idx / 169;
    const float* src = f3 + (size_t)bc * 14 * 14;
    float s = 0.f;
    #pragma unroll
    for (int dy = 0; dy < 4; ++dy) {
        int yy = y - 1 + dy;
        if (yy < 0 || yy >= 14) continue;
        #pragma unroll
        for (int dx = 0; dx < 4; ++dx) {
            int xx = x - 1 + dx;
            if (xx < 0 || xx >= 14) continue;
            s += src[yy * 14 + xx];
        }
    }
    f3p[idx] = s * (1.f / 16.f);
}

// ---- 2. fused avgpool(f2) + bilinear-up(f3p) + concat -> bf16 embedding ----
__global__ __launch_bounds__(256) void k_build_emb(const float* __restrict__ f2,
                                                   const float* __restrict__ f3p,
                                                   unsigned short* __restrict__ embBF) {
    int idx = blockIdx.x * 256 + threadIdx.x;          // r*1536 + c
    const int total = ROWSP * KC;
    if (idx >= total) return;
    int c = idx % KC, r = idx / KC;
    float val = 0.f;
    if (r < ROWS) {
        int b = r / PP, p = r % PP, h = p / HP, w = p % HP;
        if (c < 512) {
            const float* src = f2 + ((size_t)b * 512 + c) * 28 * 28;
            float s = 0.f;
            #pragma unroll
            for (int dy = 0; dy < 4; ++dy) {
                int yy = h - 1 + dy;
                if (yy < 0 || yy >= 28) continue;
                #pragma unroll
                for (int dx = 0; dx < 4; ++dx) {
                    int xx = w - 1 + dx;
                    if (xx < 0 || xx >= 28) continue;
                    s += src[yy * 28 + xx];
                }
            }
            val = s * (1.f / 16.f);
        } else {
            int cc = c - 512;
            const float* src = f3p + ((size_t)b * 1024 + cc) * 169;
            float sy = (h + 0.5f) * (13.f / 27.f) - 0.5f;
            float sx = (w + 0.5f) * (13.f / 27.f) - 0.5f;
            int y0 = (int)floorf(sy), x0 = (int)floorf(sx);
            float fy = sy - y0, fx = sx - x0;
            int y0c = min(max(y0, 0), 12), y1c = min(max(y0 + 1, 0), 12);
            int x0c = min(max(x0, 0), 12), x1c = min(max(x0 + 1, 0), 12);
            float a = src[y0c * 13 + x0c], bb = src[y0c * 13 + x1c];
            float cg = src[y1c * 13 + x0c], d = src[y1c * 13 + x1c];
            val = (a * (1.f - fx) + bb * fx) * (1.f - fy) +
                  (cg * (1.f - fx) + d  * fx) * fy;
        }
    }
    embBF[idx] = f2bf(val);
}

// ---- 3. row norms of bf16 embedding ----------------------------------------
__global__ __launch_bounds__(256) void k_x2(const unsigned short* __restrict__ embBF,
                                            float* __restrict__ x2) {
    __shared__ float red[256];
    int row = blockIdx.x, tid = threadIdx.x;
    const unsigned short* p = embBF + (size_t)row * KC;
    float s = 0.f;
    for (int c = tid; c < KC; c += 256) { float f = bf2f(p[c]); s = fmaf(f, f, s); }
    red[tid] = s; __syncthreads();
    for (int o = 128; o > 0; o >>= 1) {
        if (tid < o) red[tid] += red[tid + o];
        __syncthreads();
    }
    if (tid == 0) x2[row] = red[0];
}

// ---- 4. memory bank -> bf16 + row norms ------------------------------------
__global__ __launch_bounds__(256) void k_bank_cvt(const float* __restrict__ bank,
                                                  unsigned short* __restrict__ bankBF,
                                                  float* __restrict__ y2) {
    __shared__ float red[256];
    int row = blockIdx.x, tid = threadIdx.x;
    const float* p = bank + (size_t)row * KC;
    unsigned short* q = bankBF + (size_t)row * KC;
    float s = 0.f;
    for (int c = tid; c < KC; c += 256) {
        unsigned short h = f2bf(p[c]);
        q[c] = h;
        float f = bf2f(h);
        s = fmaf(f, f, s);
    }
    red[tid] = s; __syncthreads();
    for (int o = 128; o > 0; o >>= 1) {
        if (tid < o) red[tid] += red[tid + o];
        __syncthreads();
    }
    if (tid == 0) y2[row] = red[0];
}

// ---- 5. MAIN: WMMA bf16 GEMM + fused min/argmin ----------------------------
// grid = (92 M-tiles, 5 N-segments), block = 256 (8 waves of 32).
// Waves: 4 sub-M x 2 sub-N. Each wave computes a 16x32 slab per N-step
// (A-fragment reused for two WMMAs), K-loop is register double-buffered so
// fragment loads for step k+32 overlap the WMMAs of step k.
// A (64x1536 bf16) staged in dynamic LDS (~193 KB of the 320 KB/WGP),
// padded stride keeps the ds_load_b128 fragment reads bank-conflict-free.
__global__ __launch_bounds__(256) void k_nn(const unsigned short* __restrict__ embBF,
                                            const unsigned short* __restrict__ bankBF,
                                            const float* __restrict__ x2,
                                            const float* __restrict__ y2,
                                            float* __restrict__ pVal,
                                            int* __restrict__ pIdx) {
    extern __shared__ __align__(16) char smem[];
    unsigned short* As = (unsigned short*)smem;                        // 64*1544*2
    float* sPV = (float*)(smem + 64 * LDS_STRIDE * 2);                 // 2*64 f
    int*   sPI = (int*)  (smem + 64 * LDS_STRIDE * 2 + 2 * 64 * 4);    // 2*64 i

    const int tid = threadIdx.x;
    const int tileBase = blockIdx.x * 64;
    const int seg = blockIdx.y;

    // stage A tile (64 rows x 1536 bf16) into padded LDS, 16B chunks
    for (int i = tid; i < 64 * 192; i += 256) {
        int row = i / 192, kcol = (i % 192) * 8;
        u32x4 d = *(const u32x4*)(embBF + (size_t)(tileBase + row) * KC + kcol);
        *(u32x4*)(As + row * LDS_STRIDE + kcol) = d;
    }
    __syncthreads();

    const int wave = tid >> 5, lane = tid & 31;
    const int subM = wave & 3, subN = wave >> 2;
    const int lhalf = lane >> 4;          // K-half selector per WMMA layout
    const int l15 = lane & 15;
    const int koff = lhalf * 8;           // per-lane K byte-group offset

    float x2v[8];
    #pragma unroll
    for (int v = 0; v < 8; ++v)
        x2v[v] = x2[tileBase + subM * 16 + v + lhalf * 8];

    float best[8]; int bidx[8];
    #pragma unroll
    for (int v = 0; v < 8; ++v) { best[v] = 3.4e38f; bidx[v] = 0; }

    const unsigned short* Abase = As + (subM * 16 + l15) * LDS_STRIDE + koff;
    const int segStart = seg * SEGLEN, segEnd = segStart + SEGLEN;

    // each wave: 32 N-columns per step; the two sub-N waves interleave => 64/step
    for (int n0 = segStart + subN * 32; n0 < segEnd; n0 += 64) {
        const int col0 = n0 + l15;           // columns 0..15 of this wave's slab
        const int col1 = n0 + 16 + l15;      // columns 16..31
        const float y2l0 = y2[col0];
        const float y2l1 = y2[col1];
        const unsigned short* B0 = bankBF + (size_t)col0 * KC + koff;
        const unsigned short* B1 = bankBF + (size_t)col1 * KC + koff;
        if (n0 + 64 < segEnd) {
            __builtin_prefetch(bankBF + (size_t)(col0 + 64) * KC, 0, 1);
            __builtin_prefetch(bankBF + (size_t)(col1 + 64) * KC, 0, 1);
        }

        f32x8 acc0 = {0.f, 0.f, 0.f, 0.f, 0.f, 0.f, 0.f, 0.f};
        f32x8 acc1 = {0.f, 0.f, 0.f, 0.f, 0.f, 0.f, 0.f, 0.f};

        // -------- software-pipelined K loop (double-buffered fragments) -----
        FragAB fa, fb0, fb1;
        fa.q[0]  = *(const u32x4*)(Abase);
        fa.q[1]  = *(const u32x4*)(Abase + 16);
        fb0.q[0] = *(const u32x4*)(B0);
        fb0.q[1] = *(const u32x4*)(B0 + 16);
        fb1.q[0] = *(const u32x4*)(B1);
        fb1.q[1] = *(const u32x4*)(B1 + 16);

        #pragma unroll 4
        for (int k0 = 32; k0 < KC; k0 += 32) {
            FragAB na, nb0, nb1;                      // next-step fragments
            na.q[0]  = *(const u32x4*)(Abase + k0);
            na.q[1]  = *(const u32x4*)(Abase + k0 + 16);
            nb0.q[0] = *(const u32x4*)(B0 + k0);
            nb0.q[1] = *(const u32x4*)(B0 + k0 + 16);
            nb1.q[0] = *(const u32x4*)(B1 + k0);
            nb1.q[1] = *(const u32x4*)(B1 + k0 + 16);

            acc0 = __builtin_amdgcn_wmma_f32_16x16x32_bf16(
                false, fa.v, false, fb0.v, (short)0, acc0, false, false);
            acc1 = __builtin_amdgcn_wmma_f32_16x16x32_bf16(
                false, fa.v, false, fb1.v, (short)0, acc1, false, false);

            fa = na; fb0 = nb0; fb1 = nb1;
        }
        acc0 = __builtin_amdgcn_wmma_f32_16x16x32_bf16(
            false, fa.v, false, fb0.v, (short)0, acc0, false, false);
        acc1 = __builtin_amdgcn_wmma_f32_16x16x32_bf16(
            false, fa.v, false, fb1.v, (short)0, acc1, false, false);

        // -------- fused distance + running min/argmin -----------------------
        #pragma unroll
        for (int v = 0; v < 8; ++v) {
            float d0 = x2v[v] + y2l0 - 2.0f * acc0[v];
            float d1 = x2v[v] + y2l1 - 2.0f * acc1[v];
            if (d0 < best[v]) { best[v] = d0; bidx[v] = col0; }
            if (d1 < best[v]) { best[v] = d1; bidx[v] = col1; }
        }
    }

    // cross-lane min within each 16-lane half (N spread across lanes)
    #pragma unroll
    for (int v = 0; v < 8; ++v) {
        #pragma unroll
        for (int off = 8; off >= 1; off >>= 1) {
            float ov = __shfl_xor(best[v], off, 32);
            int   oi = __shfl_xor(bidx[v], off, 32);
            if (ov < best[v]) { best[v] = ov; bidx[v] = oi; }
        }
    }
    if (l15 == 0) {
        #pragma unroll
        for (int v = 0; v < 8; ++v) {
            int lr = subM * 16 + v + lhalf * 8;     // local M row 0..63
            sPV[subN * 64 + lr] = best[v];
            sPI[subN * 64 + lr] = bidx[v];
        }
    }
    __syncthreads();
    if (tid < 64) {
        float v0 = sPV[tid], v1 = sPV[64 + tid];
        int   i0 = sPI[tid], i1 = sPI[64 + tid];
        float bv = v0; int bi = i0;
        if (v1 < bv) { bv = v1; bi = i1; }
        int m = tileBase + tid;
        pVal[(size_t)seg * ROWSP + m] = bv;
        pIdx[(size_t)seg * ROWSP + m] = bi;
    }
}

// ---- 5b. combine N-segments -> patch_scores (sqrt) + locations -------------
__global__ __launch_bounds__(256) void k_nn_combine(const float* __restrict__ pVal,
                                                    const int* __restrict__ pIdx,
                                                    float* __restrict__ ps,
                                                    int* __restrict__ loc) {
    int m = blockIdx.x * 256 + threadIdx.x;
    if (m >= ROWSP) return;
    float bv = pVal[m]; int bi = pIdx[m];
    for (int s = 1; s < NSEG; ++s) {
        float v = pVal[(size_t)s * ROWSP + m];
        if (v < bv) { bv = v; bi = pIdx[(size_t)s * ROWSP + m]; }
    }
    ps[m] = sqrtf(fmaxf(bv, 0.f));
    loc[m] = bi;
}

// ---- 6a. per-batch argmax of patch scores ----------------------------------
__global__ __launch_bounds__(256) void k_argmax(const float* __restrict__ ps,
                                                const int* __restrict__ loc,
                                                float* __restrict__ bestS,
                                                int* __restrict__ bestP,
                                                int* __restrict__ nnIdx) {
    __shared__ float sv[256]; __shared__ int si[256];
    int b = blockIdx.x, tid = threadIdx.x;
    float mv = -3.4e38f; int mi = 0;
    for (int p = tid; p < PP; p += 256) {
        float v = ps[b * PP + p];
        if (v > mv) { mv = v; mi = p; }
    }
    sv[tid] = mv; si[tid] = mi; __syncthreads();
    for (int o = 128; o > 0; o >>= 1) {
        if (tid < o && sv[tid + o] > sv[tid]) { sv[tid] = sv[tid + o]; si[tid] = si[tid + o]; }
        __syncthreads();
    }
    if (tid == 0) {
        bestS[b] = sv[0];
        bestP[b] = si[0];
        nnIdx[b] = loc[b * PP + si[0]];
    }
}

// ---- 6b. d_bank: dist(bank[nn[b]], bank)^2 for all 8 batches ---------------
__global__ __launch_bounds__(256) void k_dbank(const float* __restrict__ bank,
                                               const float* __restrict__ y2,
                                               const int* __restrict__ nnIdx,
                                               float* __restrict__ dbank) {
    __shared__ float nnrows[NB * KC];   // 48 KB
    int tid = threadIdx.x;
    for (int j = tid; j < NB * KC; j += 256) {
        int bb = j / KC, c = j % KC;
        nnrows[j] = bank[(size_t)nnIdx[bb] * KC + c];
    }
    __syncthreads();
    int n = blockIdx.x * 256 + tid;     // 32000 = 125*256 exact
    float acc[NB];
    #pragma unroll
    for (int bb = 0; bb < NB; ++bb) acc[bb] = 0.f;
    const float* row = bank + (size_t)n * KC;
    for (int k = 0; k < KC; ++k) {
        float bv = row[k];
        #pragma unroll
        for (int bb = 0; bb < NB; ++bb)
            acc[bb] = fmaf(bv, nnrows[bb * KC + k], acc[bb]);
    }
    float y2n = y2[n];
    #pragma unroll
    for (int bb = 0; bb < NB; ++bb)
        dbank[(size_t)bb * MBANK + n] = y2[nnIdx[bb]] + y2n - 2.f * acc[bb];
}

// ---- 6c. top-9 smallest (iterative min + mask) -----------------------------
__global__ __launch_bounds__(256) void k_topk(float* __restrict__ dbank,
                                              int* __restrict__ support) {
    __shared__ float sv[256]; __shared__ int si[256];
    int b = blockIdx.x, tid = threadIdx.x;
    float* row = dbank + (size_t)b * MBANK;
    for (int it = 0; it < 9; ++it) {
        float mv = 3.4e38f; int mi = 0;
        for (int n = tid; n < MBANK; n += 256) {
            float v = row[n];
            if (v < mv) { mv = v; mi = n; }
        }
        sv[tid] = mv; si[tid] = mi; __syncthreads();
        for (int o = 128; o > 0; o >>= 1) {
            if (tid < o && sv[tid + o] < sv[tid]) { sv[tid] = sv[tid + o]; si[tid] = si[tid + o]; }
            __syncthreads();
        }
        if (tid == 0) {
            support[b * 9 + it] = si[0];
            row[si[0]] = 3.4e38f;
        }
        __syncthreads();
    }
}

// ---- 6d. re-weighted anomaly score -----------------------------------------
__global__ __launch_bounds__(256) void k_score(const unsigned short* __restrict__ embBF,
                                               const float* __restrict__ bank,
                                               const float* __restrict__ bestS,
                                               const int* __restrict__ bestP,
                                               const int* __restrict__ support,
                                               float* __restrict__ outPred) {
    __shared__ float red[256]; __shared__ float dist[9];
    int b = blockIdx.x, tid = threadIdx.x;
    int r = b * PP + bestP[b];
    int sup[9];
    #pragma unroll
    for (int k = 0; k < 9; ++k) sup[k] = support[b * 9 + k];
    float acc[9];
    #pragma unroll
    for (int k = 0; k < 9; ++k) acc[k] = 0.f;
    for (int c = tid; c < KC; c += 256) {
        float f = bf2f(embBF[(size_t)r * KC + c]);
        #pragma unroll
        for (int k = 0; k < 9; ++k) {
            float d = f - bank[(size_t)sup[k] * KC + c];
            acc[k] = fmaf(d, d, acc[k]);
        }
    }
    for (int k = 0; k < 9; ++k) {
        red[tid] = acc[k]; __syncthreads();
        for (int o = 128; o > 0; o >>= 1) {
            if (tid < o) red[tid] += red[tid + o];
            __syncthreads();
        }
        if (tid == 0) dist[k] = sqrtf(fmaxf(red[0], 0.f));
        __syncthreads();
    }
    if (tid == 0) {
        float s = bestS[b], wsum = 0.f;
        for (int k = 0; k < 9; ++k) wsum += expf(dist[k]);
        outPred[b] = (1.f - expf(s) / wsum) * s;
    }
}

// ---- 7. anomaly map: resize 27->224 + 33-tap separable gaussian ------------
__global__ void k_gw(float* __restrict__ gw) {
    if (blockIdx.x == 0 && threadIdx.x == 0) {
        float w[33], s = 0.f;
        for (int i = 0; i < 33; ++i) {
            float x = (i - 16) * 0.25f;      // sigma = 4
            w[i] = expf(-0.5f * x * x);
            s += w[i];
        }
        for (int i = 0; i < 33; ++i) gw[i] = w[i] / s;
    }
}

__global__ __launch_bounds__(256) void k_resize(const float* __restrict__ ps,
                                                float* __restrict__ amr) {
    int idx = blockIdx.x * 256 + threadIdx.x;
    const int total = NB * 224 * 224;
    if (idx >= total) return;
    int x = idx % 224, y = (idx / 224) % 224, b = idx / (224 * 224);
    float sy = (y + 0.5f) * (27.f / 224.f) - 0.5f;
    float sx = (x + 0.5f) * (27.f / 224.f) - 0.5f;
    int y0 = (int)floorf(sy), x0 = (int)floorf(sx);
    float fy = sy - y0, fx = sx - x0;
    int y0c = min(max(y0, 0), 26), y1c = min(max(y0 + 1, 0), 26);
    int x0c = min(max(x0, 0), 26), x1c = min(max(x0 + 1, 0), 26);
    const float* src = ps + b * PP;
    float a = src[y0c * HP + x0c], bb = src[y0c * HP + x1c];
    float c = src[y1c * HP + x0c], d = src[y1c * HP + x1c];
    amr[idx] = (a * (1.f - fx) + bb * fx) * (1.f - fy) +
               (c * (1.f - fx) + d  * fx) * fy;
}

__global__ __launch_bounds__(256) void k_vblur(const float* __restrict__ amr,
                                               const float* __restrict__ gw,
                                               float* __restrict__ amt) {
    __shared__ float w[33];
    if (threadIdx.x < 33) w[threadIdx.x] = gw[threadIdx.x];
    __syncthreads();
    int idx = blockIdx.x * 256 + threadIdx.x;
    const int total = NB * 224 * 224;
    if (idx >= total) return;
    int x = idx % 224, y = (idx / 224) % 224, b = idx / (224 * 224);
    const float* src = amr + (size_t)b * 224 * 224;
    float s = 0.f;
    for (int i = 0; i < 33; ++i) {
        int yy = reflect_idx(y - 16 + i, 224);
        s = fmaf(w[i], src[yy * 224 + x], s);
    }
    amt[idx] = s;
}

__global__ __launch_bounds__(256) void k_hblur(const float* __restrict__ amt,
                                               const float* __restrict__ gw,
                                               float* __restrict__ outMap) {
    __shared__ float w[33];
    if (threadIdx.x < 33) w[threadIdx.x] = gw[threadIdx.x];
    __syncthreads();
    int idx = blockIdx.x * 256 + threadIdx.x;
    const int total = NB * 224 * 224;
    if (idx >= total) return;
    int x = idx % 224, y = (idx / 224) % 224, b = idx / (224 * 224);
    const float* src = amt + (size_t)b * 224 * 224;
    float s = 0.f;
    for (int i = 0; i < 33; ++i) {
        int xx = reflect_idx(x - 16 + i, 224);
        s = fmaf(w[i], src[y * 224 + xx], s);
    }
    outMap[idx] = s;
}

// ---------------------------------------------------------------------------
// workspace layout (bytes, 256-aligned)
// ---------------------------------------------------------------------------
static inline size_t alignup(size_t x) { return (x + 255) & ~(size_t)255; }

extern "C" void kernel_launch(void* const* d_in, const int* in_sizes, int n_in,
                              void* d_out, int out_size, void* d_ws, size_t ws_size,
                              hipStream_t stream) {
    (void)in_sizes; (void)n_in; (void)out_size; (void)ws_size;
    const float* f2   = (const float*)d_in[0];   // [8,512,28,28]
    const float* f3   = (const float*)d_in[1];   // [8,1024,14,14]
    const float* bank = (const float*)d_in[2];   // [32000,1536]
    float* outMap  = (float*)d_out;              // [8,1,224,224]
    float* outPred = (float*)d_out + NB * 224 * 224;  // [8]

    char* ws = (char*)d_ws;
    size_t off = 0;
    float*          f3p    = (float*)(ws + off);          off = alignup(off + (size_t)NB*1024*169*4);
    unsigned short* embBF  = (unsigned short*)(ws + off); off = alignup(off + (size_t)ROWSP*KC*2);
    unsigned short* bankBF = (unsigned short*)(ws + off); off = alignup(off + (size_t)MBANK*KC*2);
    float*          x2     = (float*)(ws + off);          off = alignup(off + (size_t)ROWSP*4);
    float*          y2     = (float*)(ws + off);          off = alignup(off + (size_t)MBANK*4);
    float*          pVal   = (float*)(ws + off);          off = alignup(off + (size_t)NSEG*ROWSP*4);
    int*            pIdx   = (int*)(ws + off);            off = alignup(off + (size_t)NSEG*ROWSP*4);
    float*          ps     = (float*)(ws + off);          off = alignup(off + (size_t)ROWSP*4);
    int*            loc    = (int*)(ws + off);            off = alignup(off + (size_t)ROWSP*4);
    float*          dbank  = (float*)(ws + off);          off = alignup(off + (size_t)NB*MBANK*4);
    float*          bestS  = (float*)(ws + off);          off = alignup(off + NB*4);
    int*            bestP  = (int*)(ws + off);            off = alignup(off + NB*4);
    int*            nnIdx  = (int*)(ws + off);            off = alignup(off + NB*4);
    int*            support= (int*)(ws + off);            off = alignup(off + NB*9*4);
    float*          gw     = (float*)(ws + off);          off = alignup(off + 33*4);
    float*          amr    = (float*)(ws + off);          off = alignup(off + (size_t)NB*224*224*4);
    float*          amt    = (float*)(ws + off);          off = alignup(off + (size_t)NB*224*224*4);

    // big dynamic LDS for the WMMA kernel (A tile + partial min arrays)
    const int SMEM = 64 * LDS_STRIDE * 2 + 2 * 64 * 4 + 2 * 64 * 4;   // ~198.7 KB
    hipFuncSetAttribute(reinterpret_cast<const void*>(k_nn),
                        hipFuncAttributeMaxDynamicSharedMemorySize, SMEM);

    // 1-2: pooled features + bf16 embedding
    k_pool_f3<<<(NB*1024*169 + 255)/256, 256, 0, stream>>>(f3, f3p);
    k_build_emb<<<(ROWSP*KC)/256, 256, 0, stream>>>(f2, f3p, embBF);
    // 3-4: norms + bank conversion
    k_x2<<<ROWSP, 256, 0, stream>>>(embBF, x2);
    k_bank_cvt<<<MBANK, 256, 0, stream>>>(bank, bankBF, y2);
    // 5: WMMA distance search + combine
    k_nn<<<dim3(ROWSP/64, NSEG), 256, SMEM, stream>>>(embBF, bankBF, x2, y2, pVal, pIdx);
    k_nn_combine<<<(ROWSP + 255)/256, 256, 0, stream>>>(pVal, pIdx, ps, loc);
    // 6: anomaly-score re-weighting branch
    k_argmax<<<NB, 256, 0, stream>>>(ps, loc, bestS, bestP, nnIdx);
    k_dbank<<<MBANK/256, 256, 0, stream>>>(bank, y2, nnIdx, dbank);
    k_topk<<<NB, 256, 0, stream>>>(dbank, support);
    k_score<<<NB, 256, 0, stream>>>(embBF, bank, bestS, bestP, support, outPred);
    // 7: anomaly map
    k_gw<<<1, 64, 0, stream>>>(gw);
    k_resize<<<(NB*224*224 + 255)/256, 256, 0, stream>>>(ps, amr);
    k_vblur<<<(NB*224*224 + 255)/256, 256, 0, stream>>>(amr, gw, amt);
    k_hblur<<<(NB*224*224 + 255)/256, 256, 0, stream>>>(amt, gw, outMap);
}
